// CircleLoss_26792005992922
// MI455X (gfx1250) — matrix operand order repaired and checked by reading, not compiled, not run on hardware
//
#include <hip/hip_runtime.h>
#include <cstdint>
#include <cstddef>

// ---------------------------------------------------------------------------
// CircleLoss fused kernel set for gfx1250 (MI455X, wave32, WMMA).
//   sim = normalize(E) @ normalize(E)^T   via bf16 hi/lo split (3 WMMAs/step)
//   fused circle transform + online masked logsumexp, no sim materialization.
//   B operand staged LDS-side with async global->LDS DMA (ASYNCcnt path),
//   double buffered so DMA overlaps the WMMA pipeline.
// ---------------------------------------------------------------------------

typedef __bf16 bf16_t;
typedef __attribute__((ext_vector_type(16))) __bf16 v16bf;
typedef __attribute__((ext_vector_type(8)))  float  v8f;

static constexpr int N = 4096;
static constexpr int D = 1024;
static constexpr int CSPLIT = 8;                 // column chunks (gridDim.y)
static constexpr int COLS_PER_CHUNK = N / CSPLIT; // 512
static constexpr int MTILE = 128;                // rows per block (8 waves x 16)
static constexpr int NGROUP = 128;               // cols per inner group (8 n-tiles)
static constexpr int KCHUNK = 32;                // K staged per async chunk
static constexpr int BSTRIDE = KCHUNK + 8;       // padded col stride (elements) -> no LDS bank conflicts

union Frag { v16bf v; uint4 u[2]; };

__device__ __forceinline__ void lse_push(float& mx, float& s, float v) {
  if (v > mx) { s = s * __expf(mx - v) + 1.0f; mx = v; }
  else        { s += __expf(v - mx); }
}

__device__ __forceinline__ void lse_merge(float& mx, float& s, float omx, float os) {
  float M = fmaxf(mx, omx);
  float a = (s  > 0.0f) ? s  * __expf(mx  - M) : 0.0f;
  float b = (os > 0.0f) ? os * __expf(omx - M) : 0.0f;
  mx = M; s = a + b;
}

// Issue async global->LDS b128 DMA for one 128col x KCHUNK bf16 plane.
// 512 16-byte pieces, 2 per thread; consecutive threads -> consecutive 16B
// pieces of the same column (coalesced 64B/4 threads on the global side).
__device__ __forceinline__ void stage_async(const bf16_t* __restrict__ src,
                                            uint32_t ldsPlane, int nb, int k0, int tid) {
  #pragma unroll
  for (int q = 0; q < 2; ++q) {
    const int i   = tid + q * 256;        // 0..511
    const int col = i >> 2;               // 0..127
    const int kp  = (i & 3) * 8;          // element offset 0,8,16,24
    const uint64_t ga = (uint64_t)(uintptr_t)(src + (size_t)(nb + col) * D + (size_t)(k0 + kp));
    const uint32_t la = ldsPlane + (uint32_t)(col * BSTRIDE + kp) * 2u;
    asm volatile("global_load_async_to_lds_b128 %0, %1, off"
                 :: "v"(la), "v"(ga) : "memory");
  }
}

__device__ __forceinline__ void wait_async_all() {
  asm volatile("s_wait_asynccnt 0" ::: "memory");
}

// ---------------------------------------------------------------------------
// Kernel 1: row-wise L2 normalize + split into bf16 hi/lo pair.
// ---------------------------------------------------------------------------
__global__ __launch_bounds__(256) void normalize_split_kernel(
    const float* __restrict__ emb, bf16_t* __restrict__ ehi, bf16_t* __restrict__ elo) {
  const int row = blockIdx.x;
  const float4* rp = reinterpret_cast<const float4*>(emb + (size_t)row * D);
  float4 v = rp[threadIdx.x];
  float ss = v.x * v.x + v.y * v.y + v.z * v.z + v.w * v.w;

  #pragma unroll
  for (int off = 16; off > 0; off >>= 1) ss += __shfl_xor(ss, off, 32);

  __shared__ float wsum[8];
  const int wave = threadIdx.x >> 5;
  const int lane = threadIdx.x & 31;
  if (lane == 0) wsum[wave] = ss;
  __syncthreads();
  float tot = 0.0f;
  #pragma unroll
  for (int i = 0; i < 8; ++i) tot += wsum[i];

  const float inv = rsqrtf(fmaxf(tot, 1e-12f));
  const size_t base = (size_t)row * D + (size_t)threadIdx.x * 4;
  float xs[4] = {v.x, v.y, v.z, v.w};
  #pragma unroll
  for (int i = 0; i < 4; ++i) {
    float x = xs[i] * inv;
    bf16_t h = (bf16_t)x;
    ehi[base + i] = h;
    elo[base + i] = (bf16_t)(x - (float)h);
  }
}

// ---------------------------------------------------------------------------
// Kernel 2: fused GEMM (bf16x2 split via v_wmma_f32_16x16x32_bf16) with LDS
// async staging of the B operand + circle-loss epilogue (online masked LSE).
// Grid: (N/MTILE, CSPLIT), block 256 = 8 waves; each wave owns 16 rows.
// ---------------------------------------------------------------------------
__global__ __launch_bounds__(256) void circle_partial_kernel(
    const bf16_t* __restrict__ ehi, const bf16_t* __restrict__ elo,
    const int* __restrict__ posm, const int* __restrict__ negm,
    float* __restrict__ partials) {
  // [buf][hi/lo][col * BSTRIDE + k]  -> 2*2*128*40*2B = 40 KB
  __shared__ bf16_t smB[2][2][NGROUP * BSTRIDE];

  const int tid    = threadIdx.x;
  const int lane   = tid & 31;
  const int wave   = tid >> 5;
  const int l15    = lane & 15;
  const int hiHalf = lane >> 4;                   // 0 or 1
  const int mbase  = blockIdx.x * MTILE + wave * 16;
  const int cbase  = blockIdx.y * COLS_PER_CHUNK;

  const uint32_t smBase = (uint32_t)(uintptr_t)&smB[0][0][0];
  const uint32_t planeBytes = (uint32_t)(NGROUP * BSTRIDE * 2);

  // A-fragment lane addressing (ISA 16-bit 16x32 A layout):
  //   lanes 0-15: K = {0..7, 16..23}; lanes 16-31: K = {8..15, 24..31}
  const int arow  = mbase + l15;
  const int akoff = hiHalf ? 8 : 0;
  const bf16_t* aRowHi = ehi + (size_t)arow * D;
  const bf16_t* aRowLo = elo + (size_t)arow * D;

  // B-fragment lane addressing (32x16 B): lane n = lane&15, same K order as A;
  // lanes 0-15: K = {0..7, 16..23}; lanes 16-31: K = {8..15, 24..31}.
  const int bkoff = hiHalf ? 8 : 0;

  // per-row online LSE state (rows m = mbase + r + 8*hiHalf)
  float mxp[8], sp[8], cp[8], mxn[8], sn[8], cn[8];
  #pragma unroll
  for (int r = 0; r < 8; ++r) {
    mxp[r] = -3.0e38f; sp[r] = 0.0f; cp[r] = 0.0f;
    mxn[r] = -3.0e38f; sn[r] = 0.0f; cn[r] = 0.0f;
  }

  for (int cg = 0; cg < COLS_PER_CHUNK / NGROUP; ++cg) {
    const int nb = cbase + cg * NGROUP;

    v8f acc[8];
    #pragma unroll
    for (int j = 0; j < 8; ++j) acc[j] = v8f{0,0,0,0,0,0,0,0};

    // prefetch chunk 0 into buffer 0
    stage_async(ehi, smBase + 0u * planeBytes, nb, 0, tid);
    stage_async(elo, smBase + 1u * planeBytes, nb, 0, tid);

    for (int kc = 0; kc < D / KCHUNK; ++kc) {
      wait_async_all();        // my async issues done
      __syncthreads();         // everyone's staged data visible; prev buffer free
      const int nxt = kc + 1;
      if (nxt < D / KCHUNK) {
        const uint32_t bufOff = (uint32_t)(nxt & 1) * 2u * planeBytes;
        stage_async(ehi, smBase + bufOff + 0u * planeBytes, nb, nxt * KCHUNK, tid);
        stage_async(elo, smBase + bufOff + 1u * planeBytes, nb, nxt * KCHUNK, tid);
      }

      const bf16_t* sBh = &smB[kc & 1][0][0];
      const bf16_t* sBl = &smB[kc & 1][1][0];
      const int k0 = kc * KCHUNK;

      Frag ah, al;
      {
        const uint4* pah = reinterpret_cast<const uint4*>(aRowHi + k0 + akoff);
        const uint4* pal = reinterpret_cast<const uint4*>(aRowLo + k0 + akoff);
        ah.u[0] = pah[0]; ah.u[1] = pah[2];   // K+{0..7} and K+{16..23}
        al.u[0] = pal[0]; al.u[1] = pal[2];
      }
      #pragma unroll
      for (int j = 0; j < 8; ++j) {
        Frag bh, bl;
        const int colLocal = j * 16 + l15;
        const uint4* pbh = reinterpret_cast<const uint4*>(sBh + colLocal * BSTRIDE + bkoff);
        const uint4* pbl = reinterpret_cast<const uint4*>(sBl + colLocal * BSTRIDE + bkoff);
        bh.u[0] = pbh[0]; bh.u[1] = pbh[2];
        bl.u[0] = pbl[0]; bl.u[1] = pbl[2];
        // sim += hi*hi + hi*lo + lo*hi  (chained f32 accumulator)
        acc[j] = __builtin_amdgcn_wmma_f32_16x16x32_bf16(false, ah.v, false, bh.v,
                                                         (short)0, acc[j], false, false);
        acc[j] = __builtin_amdgcn_wmma_f32_16x16x32_bf16(false, ah.v, false, bl.v,
                                                         (short)0, acc[j], false, false);
        acc[j] = __builtin_amdgcn_wmma_f32_16x16x32_bf16(false, al.v, false, bh.v,
                                                         (short)0, acc[j], false, false);
      }
      // WMMA consumption forces s_wait_dscnt before each thread reaches the
      // next barrier, so the buffer can be safely overwritten next round.
    }

    // Fused circle-loss epilogue. C layout: n = lane&15, m = r + 8*(lane>>4).
    #pragma unroll
    for (int j = 0; j < 8; ++j) {
      const int n = nb + j * 16 + l15;
      #pragma unroll
      for (int r = 0; r < 8; ++r) {
        const int m = mbase + r + 8 * hiHalf;
        const size_t idx = (size_t)m * N + (size_t)n;
        const float v  = acc[j][r];
        const float ap = -256.0f * fmaxf(1.25f - v, 0.0f) * (v - 0.75f);
        const float an =  256.0f * fmaxf(v + 0.25f, 0.0f) * (v - 0.25f);
        const int p = posm[idx];
        const int g = negm[idx];
        const float nm = g ? an : (p ? ap : 0.0f);   // neg overwrites overlaps
        if (p) { lse_push(mxp[r], sp[r], nm); cp[r] += 1.0f; }
        if (g) { lse_push(mxn[r], sn[r], nm); cn[r] += 1.0f; }
      }
    }
  }

  // Reduce each row's state across the 16 lanes of its half.
  #pragma unroll
  for (int r = 0; r < 8; ++r) {
    #pragma unroll
    for (int off = 1; off < 16; off <<= 1) {
      float omx = __shfl_xor(mxp[r], off, 32);
      float os  = __shfl_xor(sp[r],  off, 32);
      float oc  = __shfl_xor(cp[r],  off, 32);
      lse_merge(mxp[r], sp[r], omx, os); cp[r] += oc;
      omx = __shfl_xor(mxn[r], off, 32);
      os  = __shfl_xor(sn[r],  off, 32);
      oc  = __shfl_xor(cn[r],  off, 32);
      lse_merge(mxn[r], sn[r], omx, os); cn[r] += oc;
    }
    if (l15 == 0) {
      const int row = mbase + r + 8 * hiHalf;
      float* o = partials + ((size_t)blockIdx.y * N + (size_t)row) * 6;
      o[0] = mxp[r]; o[1] = sp[r]; o[2] = cp[r];
      o[3] = mxn[r]; o[4] = sn[r]; o[5] = cn[r];
    }
  }
}

// ---------------------------------------------------------------------------
// Kernel 3: merge column-chunk partials per row, softplus, validity, mean.
// ---------------------------------------------------------------------------
__global__ __launch_bounds__(256) void finalize_kernel(
    const float* __restrict__ partials, float* __restrict__ out) {
  float lsum = 0.0f, lcnt = 0.0f;
  for (int row = threadIdx.x; row < N; row += 256) {
    float mxp = -3.0e38f, sp = 0.0f, cp = 0.0f;
    float mxn = -3.0e38f, sn = 0.0f, cn = 0.0f;
    for (int cs = 0; cs < CSPLIT; ++cs) {
      const float* p = partials + ((size_t)cs * N + (size_t)row) * 6;
      lse_merge(mxp, sp, p[0], p[1]); cp += p[2];
      lse_merge(mxn, sn, p[3], p[4]); cn += p[5];
    }
    const float lp = (sp > 0.0f) ? (logf(sp) + mxp) : 0.0f;  // empty mask -> lse 0
    const float ln = (sn > 0.0f) ? (logf(sn) + mxn) : 0.0f;
    const float t = lp + ln;
    float loss = fmaxf(t, 0.0f) + log1pf(expf(-fabsf(t)));   // stable softplus
    const bool valid = (cp > 0.0f) && (cn > 0.0f);
    loss = valid ? loss : 0.0f;
    if (loss > 0.0f) { lsum += loss; lcnt += 1.0f; }
  }
  __shared__ float ssum[256];
  __shared__ float scnt[256];
  ssum[threadIdx.x] = lsum; scnt[threadIdx.x] = lcnt;
  __syncthreads();
  for (int st = 128; st > 0; st >>= 1) {
    if (threadIdx.x < st) {
      ssum[threadIdx.x] += ssum[threadIdx.x + st];
      scnt[threadIdx.x] += scnt[threadIdx.x + st];
    }
    __syncthreads();
  }
  if (threadIdx.x == 0) out[0] = (scnt[0] > 0.0f) ? (ssum[0] / scnt[0]) : 0.0f;
}

// ---------------------------------------------------------------------------
extern "C" void kernel_launch(void* const* d_in, const int* in_sizes, int n_in,
                              void* d_out, int out_size, void* d_ws, size_t ws_size,
                              hipStream_t stream) {
  (void)in_sizes; (void)n_in; (void)out_size; (void)ws_size;
  const float* emb = (const float*)d_in[0];
  const int*   pos = (const int*)d_in[1];
  const int*   neg = (const int*)d_in[2];
  float* out = (float*)d_out;

  // workspace layout: e_hi (8MB) | e_lo (8MB) | partials (CSPLIT*N*6 f32)
  bf16_t* ehi = (bf16_t*)d_ws;
  bf16_t* elo = ehi + (size_t)N * D;
  float* partials = (float*)(elo + (size_t)N * D);

  normalize_split_kernel<<<N, 256, 0, stream>>>(emb, ehi, elo);
  circle_partial_kernel<<<dim3(N / MTILE, CSPLIT), 256, 0, stream>>>(ehi, elo, pos, neg, partials);
  finalize_kernel<<<1, 256, 0, stream>>>(partials, out);
}
